// TransformerBlock_31190052503797
// MI455X (gfx1250) — compile-verified, matrix operand order checked by reading, not run on hardware
//
#include <hip/hip_runtime.h>

// ---------------------------------------------------------------------------
// Types for CDNA5 WMMA (wave32, 16x16x32 bf16 -> f32)
// ---------------------------------------------------------------------------
typedef __attribute__((ext_vector_type(16))) __bf16 v16bf;
typedef __attribute__((ext_vector_type(8)))  float  v8f;

union AFrag {                 // 16x32 bf16 operand fragment (8 VGPRs)
    v16bf v;
    unsigned int u[8];
    unsigned short s[16];
};

__device__ __forceinline__ unsigned short f2bf(float f) {
    unsigned int u = __float_as_uint(f);
    u += 0x7fffu + ((u >> 16) & 1u);      // round-to-nearest-even
    return (unsigned short)(u >> 16);
}

__device__ __forceinline__ v8f wmma_bf16(const AFrag& a, const AFrag& b, v8f c) {
    return __builtin_amdgcn_wmma_f32_16x16x32_bf16(
        false, a.v, false, b.v, (short)0, c, false, false);
}

// K index of VGPR-pair j for 16-bit A/B fragment layout (ISA 7.12.2):
// lanes 0-15 hold K=0..7,16..23 ; lanes 16-31 hold K=8..15,24..31 (pairs).
__device__ __forceinline__ int kpair(int j, int half) {
    return (j < 4) ? (half * 8 + j * 2) : (16 + half * 8 + (j - 4) * 2);
}

// Load a 16x32 fragment from LDS laid out [rows][stride] (stride in u16, even)
__device__ __forceinline__ AFrag ld_frag(const unsigned short* sh, int base,
                                         int stride, int l15, int half) {
    AFrag f;
    const unsigned short* p = sh + (base + l15) * stride;
#pragma unroll
    for (int j = 0; j < 8; ++j)
        f.u[j] = *(const unsigned int*)(p + kpair(j, half));
    return f;
}

// ---------------------------------------------------------------------------
// CDNA5 async global->LDS copy (ASYNCcnt path, ISA 10 / 15.18.3)
// 16 bytes per lane; LDS dest and global src must be 16B aligned.
// Generic shared pointers carry the LDS byte offset in their low 32 bits
// ---------------------------------------------------------------------------
__device__ __forceinline__ void async_g2l_b128(void* lds_ptr, const void* gptr) {
    unsigned dst = (unsigned)(unsigned long long)lds_ptr;
    asm volatile("global_load_async_to_lds_b128 %0, %1, off"
                 :: "v"(dst), "v"((unsigned long long)gptr)
                 : "memory");
}
__device__ __forceinline__ void async_wait0() {
    asm volatile("s_wait_asynccnt 0" ::: "memory");
}

// ---------------------------------------------------------------------------
// fp32 -> bf16 conversion
// ---------------------------------------------------------------------------
__global__ __launch_bounds__(256) void cvt_bf16_kernel(
    const float* __restrict__ in, unsigned short* __restrict__ out, int n) {
    int i = blockIdx.x * 256 + threadIdx.x;
    if (i < n) out[i] = f2bf(in[i]);
}

// ---------------------------------------------------------------------------
// LayerNorm (one row of 2048 per block) -> bf16
// ---------------------------------------------------------------------------
__global__ __launch_bounds__(256) void ln_bf16_kernel(
    const float* __restrict__ x, const float* __restrict__ g,
    const float* __restrict__ beta, unsigned short* __restrict__ out) {
    const int D = 2048;
    int row = blockIdx.x;
    const float* xr = x + (size_t)row * D;
    float s = 0.f, s2 = 0.f;
    for (int i = threadIdx.x; i < D; i += 256) {
        float v = xr[i];
        s += v; s2 += v * v;
    }
    __shared__ float sh[256], sh2[256];
    sh[threadIdx.x] = s; sh2[threadIdx.x] = s2;
    __syncthreads();
    for (int o = 128; o > 0; o >>= 1) {
        if (threadIdx.x < o) {
            sh[threadIdx.x]  += sh[threadIdx.x + o];
            sh2[threadIdx.x] += sh2[threadIdx.x + o];
        }
        __syncthreads();
    }
    float mean = sh[0] * (1.f / D);
    float var  = sh2[0] * (1.f / D) - mean * mean;
    float rstd = rsqrtf(var + 1e-5f);
    unsigned short* orow = out + (size_t)row * D;
    for (int i = threadIdx.x; i < D; i += 256)
        orow[i] = f2bf((xr[i] - mean) * rstd * g[i] + beta[i]);
}

// ---------------------------------------------------------------------------
// Tiled bf16 WMMA GEMM: C[M,N] = A[M,K] * B[K,N] + bias (+ epilogue)
// Block tile 128x128, 8 waves (4x2), each wave 32x64 (8 accumulators).
// A tile staged via async global->LDS (ASYNCcnt); B staged transposed via ds.
// EPI 0: fp32 out = acc + bias + res
// EPI 1: QKV scatter -> bf16 q/k/v buffers [B,H,T,64]
// EPI 2: bf16 out = gelu(acc + bias)
// ---------------------------------------------------------------------------
#define ASTRIDE 40   // u16 stride: 80B pitch (16B-aligned rows), conflict-free
template <int EPI>
__global__ __launch_bounds__(256) void gemm_bf16_kernel(
    const unsigned short* __restrict__ A, const unsigned short* __restrict__ Bw,
    const float* __restrict__ bias, const float* __restrict__ res,
    float* __restrict__ outF, unsigned short* __restrict__ outU0,
    unsigned short* __restrict__ outU1, unsigned short* __restrict__ outU2,
    int M, int N, int K) {
    __shared__ unsigned short a_sh[128 * ASTRIDE];  // [row][k]
    __shared__ unsigned short b_sh[128 * ASTRIDE];  // [col][k] (transposed)

    int tid = threadIdx.x;
    int lane = tid & 31, wave = tid >> 5;
    int half = lane >> 4, l15 = lane & 15;
    int wm = wave >> 1, wn = wave & 1;
    int m0 = blockIdx.y * 128;
    int n0 = blockIdx.x * 128;

    v8f acc[2][4];
    v8f zero = {};
#pragma unroll
    for (int i = 0; i < 2; ++i)
#pragma unroll
        for (int j = 0; j < 4; ++j) acc[i][j] = zero;

    int bn8 = (tid & 15) * 8;   // 0..120 : column-chunk base within tile
    int bks = tid >> 4;         // 0..15  : k within half-step

    for (int k0 = 0; k0 < K; k0 += 32) {
        // stage A via async global->LDS (2 x 16B per thread)
#pragma unroll
        for (int i = 0; i < 2; ++i) {
            int c = tid + i * 256;
            int row = c >> 2;
            int colb = (c & 3) * 8;
            async_g2l_b128(a_sh + row * ASTRIDE + colb,
                           A + (size_t)(m0 + row) * K + k0 + colb);
        }
        // stage B transposed: coalesced 16B reads along n, scatter to b_sh[n][k]
#pragma unroll
        for (int kk = 0; kk < 32; kk += 16) {
            int k = kk + bks;
            union { uint4 q; unsigned short w[8]; } t;
            t.q = *(const uint4*)(Bw + (size_t)(k0 + k) * N + n0 + bn8);
#pragma unroll
            for (int j = 0; j < 8; ++j)
                b_sh[(bn8 + j) * ASTRIDE + k] = t.w[j];
        }
        // prefetch next tiles into L2 while we compute (global_prefetch_b8)
        if (k0 + 32 < K) {
            __builtin_prefetch(A + (size_t)(m0 + (tid >> 1)) * K + k0 + 32, 0, 0);
            __builtin_prefetch(Bw + (size_t)(k0 + 32 + bks) * N + n0 + bn8, 0, 0);
            __builtin_prefetch(Bw + (size_t)(k0 + 48 + bks) * N + n0 + bn8, 0, 0);
        }
        async_wait0();
        __syncthreads();

        AFrag af[2], bfr[4];
#pragma unroll
        for (int i = 0; i < 2; ++i)
            af[i] = ld_frag(a_sh, wm * 32 + i * 16, ASTRIDE, l15, half);
#pragma unroll
        for (int j = 0; j < 4; ++j)
            bfr[j] = ld_frag(b_sh, wn * 64 + j * 16, ASTRIDE, l15, half);
#pragma unroll
        for (int i = 0; i < 2; ++i)
#pragma unroll
            for (int j = 0; j < 4; ++j)
                acc[i][j] = wmma_bf16(af[i], bfr[j], acc[i][j]);
        __syncthreads();
    }

    // epilogue (C layout: VGPR r -> row half*8+r, col = lane&15)
#pragma unroll
    for (int i = 0; i < 2; ++i) {
#pragma unroll
        for (int j = 0; j < 4; ++j) {
#pragma unroll
            for (int r = 0; r < 8; ++r) {
                int row = m0 + wm * 32 + i * 16 + half * 8 + r;
                int col = n0 + wn * 64 + j * 16 + l15;
                float v = acc[i][j][r] + bias[col];
                if constexpr (EPI == 0) {
                    size_t idx = (size_t)row * N + col;
                    outF[idx] = v + res[idx];
                } else if constexpr (EPI == 1) {
                    int part = col >> 11, rem = col & 2047;
                    int head = rem >> 6, d = rem & 63;
                    int bb = row >> 11, t = row & 2047;
                    size_t di = (((size_t)(bb * 32 + head)) * 2048 + t) * 64 + d;
                    unsigned short pv = f2bf(v);
                    if (part == 0)      outU0[di] = pv;
                    else if (part == 1) outU1[di] = pv;
                    else                outU2[di] = pv;
                } else {  // EPI == 2 : exact-erf GELU -> bf16
                    float gv = 0.5f * v * (1.0f + erff(v * 0.70710678f));
                    outU0[(size_t)row * N + col] = f2bf(gv);
                }
            }
        }
    }
}

// ---------------------------------------------------------------------------
// Flash attention: one (b,h) x 64-query tile per block, 4 waves,
// each wave owns 16 query rows. Keys processed 64 at a time (16 WMMA/step).
// K tile staged async global->LDS; V staged transposed; shared by all waves.
// scores = Q·K^T via WMMA, online softmax (16-lane shfl reductions),
// P re-laid-out through per-wave LDS, then P·V via WMMA.
// ---------------------------------------------------------------------------
#define KSTRIDE 72   // u16 stride: 144B pitch (16B aligned), conflict-free
#define VSTRIDE 68   // u16 stride for v_sh [d][s]: (d*34+s/2)%64 conflict-free
#define PSTRIDE 66   // u16 stride for p_sh: (r*33+s/2)%64 conflict-free
__global__ __launch_bounds__(128) void attention_kernel(
    const unsigned short* __restrict__ Q, const unsigned short* __restrict__ Kb,
    const unsigned short* __restrict__ V, unsigned short* __restrict__ att) {
    const int T = 2048, D = 64;
    int bh = blockIdx.x;                    // 0..63 = b*32 + h
    int b = bh >> 5, h = bh & 31;
    int tid = threadIdx.x, lane = tid & 31, wave = tid >> 5;
    int half = lane >> 4, l15 = lane & 15;
    int q0 = blockIdx.y * 64 + wave * 16;

    __shared__ unsigned short k_sh[64 * KSTRIDE];     // [s][d]
    __shared__ unsigned short v_sh[64 * VSTRIDE];     // [d][s] (transposed)
    __shared__ unsigned short p_sh[4][16 * PSTRIDE];  // per-wave P tile 16x64

    const unsigned short* qbase = Q + (size_t)bh * T * D;
    const unsigned short* kbase = Kb + (size_t)bh * T * D;
    const unsigned short* vbase = V + (size_t)bh * T * D;

    // Q fragments: rows q0+l15, d-chunks [0,32) and [32,64)
    AFrag qa[2];
    {
        const unsigned short* qr = qbase + (size_t)(q0 + l15) * D;
#pragma unroll
        for (int c = 0; c < 2; ++c)
#pragma unroll
            for (int j = 0; j < 8; ++j)
                qa[c].u[j] = *(const unsigned int*)(qr + c * 32 + kpair(j, half));
    }

    float mrow[8], lrow[8];
    v8f ov[4];
    v8f zero = {};
#pragma unroll
    for (int r = 0; r < 8; ++r) { mrow[r] = -1e30f; lrow[r] = 0.f; }
#pragma unroll
    for (int d = 0; d < 4; ++d) ov[d] = zero;

    const float scale = 0.125f;  // 1/sqrt(64)

    for (int s0 = 0; s0 < T; s0 += 64) {
        // stage K (async, row-major) and V (transposed via ds) tiles: 64x64
#pragma unroll
        for (int i = 0; i < 4; ++i) {
            int c = tid + i * 128;        // 0..511 chunks of 8 u16
            int s = c >> 3;               // 0..63
            int db = (c & 7) * 8;         // 0..56
            async_g2l_b128(k_sh + s * KSTRIDE + db,
                           kbase + (size_t)(s0 + s) * D + db);
            union { uint4 q; unsigned short w[8]; } tv;
            tv.q = *(const uint4*)(vbase + (size_t)(s0 + s) * D + db);
#pragma unroll
            for (int j = 0; j < 8; ++j)
                v_sh[(db + j) * VSTRIDE + s] = tv.w[j];
        }
        if (s0 + 64 < T) {
            __builtin_prefetch(kbase + (size_t)(s0 + 64 + (tid >> 1)) * D, 0, 0);
            __builtin_prefetch(vbase + (size_t)(s0 + 64 + (tid >> 1)) * D, 0, 0);
        }
        async_wait0();
        __syncthreads();

        // scores for 64 keys: 4 n-chunks of 16, K-dim = d (2 WMMA each)
        v8f sc[4];
#pragma unroll
        for (int nh = 0; nh < 4; ++nh) {
            const unsigned short* kr = k_sh + (nh * 16 + l15) * KSTRIDE;
            AFrag kf0, kf1;
#pragma unroll
            for (int j = 0; j < 8; ++j) {
                kf0.u[j] = *(const unsigned int*)(kr + kpair(j, half));
                kf1.u[j] = *(const unsigned int*)(kr + 32 + kpair(j, half));
            }
            v8f c = zero;
            c = wmma_bf16(qa[0], kf0, c);
            c = wmma_bf16(qa[1], kf1, c);
            sc[nh] = c;
        }
        // online softmax update, write P (bf16) to per-wave LDS
#pragma unroll
        for (int r = 0; r < 8; ++r) {
            float sv[4];
#pragma unroll
            for (int nh = 0; nh < 4; ++nh) sv[nh] = sc[nh][r] * scale;
            float mx = fmaxf(fmaxf(sv[0], sv[1]), fmaxf(sv[2], sv[3]));
#pragma unroll
            for (int m = 1; m < 16; m <<= 1) mx = fmaxf(mx, __shfl_xor(mx, m, 32));
            float mnew = fmaxf(mrow[r], mx);
            float alpha = __expf(mrow[r] - mnew);
            float p[4], ps = 0.f;
#pragma unroll
            for (int nh = 0; nh < 4; ++nh) { p[nh] = __expf(sv[nh] - mnew); ps += p[nh]; }
#pragma unroll
            for (int m = 1; m < 16; m <<= 1) ps += __shfl_xor(ps, m, 32);
            lrow[r] = lrow[r] * alpha + ps;
            mrow[r] = mnew;
#pragma unroll
            for (int d = 0; d < 4; ++d) ov[d][r] *= alpha;
            int row = half * 8 + r;
#pragma unroll
            for (int nh = 0; nh < 4; ++nh)
                p_sh[wave][row * PSTRIDE + nh * 16 + l15] = f2bf(p[nh]);
        }
        // P as two A-fragments (16 rows x 32 keys each)
        AFrag pa[2];
        {
            const unsigned short* pr = &p_sh[wave][l15 * PSTRIDE];
#pragma unroll
            for (int c = 0; c < 2; ++c)
#pragma unroll
                for (int j = 0; j < 8; ++j)
                    pa[c].u[j] = *(const unsigned int*)(pr + c * 32 + kpair(j, half));
        }
        // V fragments from transposed LDS (k-dim = s pairs contiguous)
#pragma unroll
        for (int dc = 0; dc < 4; ++dc) {
            const unsigned short* vr = v_sh + (dc * 16 + l15) * VSTRIDE;
            AFrag vf0, vf1;
#pragma unroll
            for (int j = 0; j < 8; ++j) {
                vf0.u[j] = *(const unsigned int*)(vr + kpair(j, half));
                vf1.u[j] = *(const unsigned int*)(vr + 32 + kpair(j, half));
            }
            ov[dc] = wmma_bf16(pa[0], vf0, ov[dc]);
            ov[dc] = wmma_bf16(pa[1], vf1, ov[dc]);
        }
        __syncthreads();
    }
    // finalize: O /= l ; write bf16 to att[b*T+t][h*64+d]
#pragma unroll
    for (int r = 0; r < 8; ++r) {
        float inv = 1.0f / lrow[r];
        int t = q0 + half * 8 + r;
        size_t rowoff = ((size_t)(b * T + t)) * 2048 + h * 64;
#pragma unroll
        for (int dc = 0; dc < 4; ++dc)
            att[rowoff + dc * 16 + l15] = f2bf(ov[dc][r] * inv);
    }
}

// ---------------------------------------------------------------------------
// Host-side orchestration
// ---------------------------------------------------------------------------
extern "C" void kernel_launch(void* const* d_in, const int* in_sizes, int n_in,
                              void* d_out, int out_size, void* d_ws, size_t ws_size,
                              hipStream_t stream) {
    (void)in_sizes; (void)n_in; (void)out_size; (void)ws_size;
    const int DIM = 2048, DFF = 8192, M = 2 * 2048, NQKV = 3 * DIM;

    const float* x     = (const float*)d_in[0];
    const float* w_qkv = (const float*)d_in[1];
    const float* b_qkv = (const float*)d_in[2];
    const float* w_o   = (const float*)d_in[3];
    const float* b_o   = (const float*)d_in[4];
    const float* g1    = (const float*)d_in[5];
    const float* beta1 = (const float*)d_in[6];
    const float* w1    = (const float*)d_in[7];
    const float* b1    = (const float*)d_in[8];
    const float* w2    = (const float*)d_in[9];
    const float* b2    = (const float*)d_in[10];
    const float* g2    = (const float*)d_in[11];
    const float* beta2 = (const float*)d_in[12];

    char* ws = (char*)d_ws;
    size_t off = 0;
    auto alloc = [&](size_t bytes) {
        void* p = ws + off;
        off += (bytes + 255) & ~(size_t)255;
        return p;
    };
    unsigned short* h1    = (unsigned short*)alloc((size_t)M * DIM * 2);
    unsigned short* wqkvb = (unsigned short*)alloc((size_t)DIM * NQKV * 2);
    unsigned short* qb    = (unsigned short*)alloc((size_t)M * DIM * 2);
    unsigned short* kb    = (unsigned short*)alloc((size_t)M * DIM * 2);
    unsigned short* vb    = (unsigned short*)alloc((size_t)M * DIM * 2);
    unsigned short* attb  = (unsigned short*)alloc((size_t)M * DIM * 2);
    unsigned short* wob   = (unsigned short*)alloc((size_t)DIM * DIM * 2);
    float*          x1    = (float*)alloc((size_t)M * DIM * 4);
    unsigned short* h2    = (unsigned short*)alloc((size_t)M * DIM * 2);
    unsigned short* w1b   = (unsigned short*)alloc((size_t)DIM * DFF * 2);
    unsigned short* ffn1  = (unsigned short*)alloc((size_t)M * DFF * 2);
    unsigned short* w2b   = (unsigned short*)alloc((size_t)DFF * DIM * 2);

    auto cvt = [&](const float* src, unsigned short* dst, int n) {
        cvt_bf16_kernel<<<(n + 255) / 256, 256, 0, stream>>>(src, dst, n);
    };
    cvt(w_qkv, wqkvb, DIM * NQKV);
    cvt(w_o,   wob,   DIM * DIM);
    cvt(w1,    w1b,   DIM * DFF);
    cvt(w2,    w2b,   DFF * DIM);

    // LN1
    ln_bf16_kernel<<<M, 256, 0, stream>>>(x, g1, beta1, h1);

    // QKV projection with scatter into q/k/v [B,H,T,64] bf16
    gemm_bf16_kernel<1><<<dim3(NQKV / 128, M / 128), 256, 0, stream>>>(
        h1, wqkvb, b_qkv, nullptr, nullptr, qb, kb, vb, M, NQKV, DIM);

    // Attention (64 bh-pairs x 32 query tiles)
    attention_kernel<<<dim3(64, 2048 / 64), 128, 0, stream>>>(qb, kb, vb, attb);

    // Output projection + residual (fp32)
    gemm_bf16_kernel<0><<<dim3(DIM / 128, M / 128), 256, 0, stream>>>(
        attb, wob, b_o, x, x1, nullptr, nullptr, nullptr, M, DIM, DIM);

    // LN2
    ln_bf16_kernel<<<M, 256, 0, stream>>>(x1, g2, beta2, h2);

    // FFN1 + exact GELU -> bf16
    gemm_bf16_kernel<2><<<dim3(DFF / 128, M / 128), 256, 0, stream>>>(
        h2, w1b, b1, nullptr, nullptr, ffn1, nullptr, nullptr, M, DFF, DIM);

    // FFN2 + residual -> d_out (fp32)
    gemm_bf16_kernel<0><<<dim3(DIM / 128, M / 128), 256, 0, stream>>>(
        ffn1, w2b, b2, x1, (float*)d_out, nullptr, nullptr, nullptr, M, DIM, DFF);
}
